// GPT_84662395338857
// MI455X (gfx1250) — matrix operand order repaired and checked by reading, not compiled, not run on hardware
//
#include <hip/hip_runtime.h>

// ---------------------------------------------------------------------------
// GPT-2-ish forward pass for MI455X (gfx1250, wave32, WMMA).
// All matmuls go through v_wmma_f32_16x16x32_bf16 (bf16 A/B, f32 accumulate):
// the problem is compute-bound (~0.83 TFLOP vs ~0.8 GB @ 23.3 TB/s), so the
// bf16 WMMA path is the right ceiling. Weights are converted f32->bf16 inside
// the GEMM's B-tile stage (uniform-branch fast path; the divergent guarded
// path only runs on the ragged last N-tile of the LM head). Next K-tile
// global data is warmed via global_prefetch_b8 one barrier-period ahead.
// ---------------------------------------------------------------------------

typedef __bf16 bf16;
typedef __attribute__((ext_vector_type(8)))  bf16  bf16x8;
typedef __attribute__((ext_vector_type(16))) bf16  bf16x16;
typedef __attribute__((ext_vector_type(8)))  float f32x8;
typedef __attribute__((ext_vector_type(4)))  float f32x4;

#define GPT_V  50257
#define GPT_D  1024
#define GPT_H  16
#define GPT_L  12
#define GPT_HS 64
#define GPT_B  8
#define GPT_T  256
#define GPT_R  (GPT_B * GPT_T) /* 2048 rows */

union FragBF { bf16x16 v; bf16x8 h[2]; };

__device__ __forceinline__ f32x8 zero_f32x8() {
  f32x8 z = {0.f, 0.f, 0.f, 0.f, 0.f, 0.f, 0.f, 0.f};
  return z;
}

__device__ __forceinline__ float blockReduceSum256(float v, float* sh) {
  const int tid = threadIdx.x;
  sh[tid] = v;
  __syncthreads();
  #pragma unroll
  for (int o = 128; o > 0; o >>= 1) {
    if (tid < o) sh[tid] += sh[tid + o];
    __syncthreads();
  }
  float r = sh[0];
  __syncthreads();
  return r;
}

__device__ __forceinline__ float blockReduceMax256(float v, float* sh) {
  const int tid = threadIdx.x;
  sh[tid] = v;
  __syncthreads();
  #pragma unroll
  for (int o = 128; o > 0; o >>= 1) {
    if (tid < o) sh[tid] = fmaxf(sh[tid], sh[tid + o]);
    __syncthreads();
  }
  float r = sh[0];
  __syncthreads();
  return r;
}

// ---------------------------------------------------------------------------
// Generic batched GEMM:  C[z] = act(A[z] (bf16, MxK) * B[z] (KxN) + bias) + resid
//  - B from bf16 (Bb) or f32-with-inline-convert (Bf)
//  - C offset for batch z: (z/zdiv)*sC0 + (z%zdiv)*sC1  (zdiv=1 -> plain batch)
//  - block tile 128x64, BK=32; 8 waves in 4x2, each wave owns 32x32 output via
//    2 A-frags x 2 B-frags -> 4 v_wmma_f32_16x16x32_bf16 per K-step.
//    A and B fragments share contiguous k order (K = half*16 + e), not the ISA table order.
//  - global_prefetch_b8 warms the next K-tile during the current one.
// ---------------------------------------------------------------------------
__global__ __launch_bounds__(256) void gemm_bf16_wmma(
    const bf16* __restrict__ A, long lda, long sAz,
    const bf16* __restrict__ Bb, const float* __restrict__ Bf, long ldb, long sBz,
    float* __restrict__ Cf, bf16* __restrict__ Cb, long ldc,
    long sC0, long sC1, int zdiv,
    const float* __restrict__ bias, const float* __restrict__ resid,
    int M, int N, int K, int relu)
{
  __shared__ __align__(16) bf16 sA[128][40];  // [m][k], pad keeps 16B alignment
  __shared__ __align__(16) bf16 sBt[64][40];  // [n][k] (B transposed on stage-in)

  const int tid  = threadIdx.x;
  const int wave = tid >> 5;
  const int lane = tid & 31;
  const int half = lane >> 4;   // lane group 0/1 (per ISA operand striping)
  const int l16  = lane & 15;
  const int mw   = wave >> 1;   // 0..3 : 32-row slab
  const int nw   = wave & 1;    // 0..1 : 32-col slab

  const int cM = blockIdx.y * 128;
  const int cN = blockIdx.x * 64;
  const int z  = blockIdx.z;

  const size_t aOff = (size_t)z * (size_t)sAz;
  const size_t bOff = (size_t)z * (size_t)sBz;
  const size_t cOff = (size_t)(z / zdiv) * (size_t)sC0 +
                      (size_t)(z % zdiv) * (size_t)sC1;

  f32x8 acc[2][2];
  acc[0][0] = zero_f32x8(); acc[0][1] = zero_f32x8();
  acc[1][0] = zero_f32x8(); acc[1][1] = zero_f32x8();

  const int bk  = tid >> 3;         // 0..31, B stage: 8 cols per thread
  const int bn8 = (tid & 7) * 8;
  const bool fullN = (cN + 64 <= N);  // block-uniform: fast path is branch-free

  for (int kt = 0; kt < K; kt += 32) {
    // ---- stage A tile (128 x 32): 2 x 16B vector loads per thread, coalesced
    #pragma unroll
    for (int c = 0; c < 2; ++c) {
      const int chunk = c * 256 + tid;     // 0..511
      const int row = chunk >> 2;          // 0..127
      const int k8  = (chunk & 3) * 8;
      const bf16* src = A + aOff + (size_t)(cM + row) * lda + (kt + k8);
      *reinterpret_cast<bf16x8*>(&sA[row][k8]) =
          *reinterpret_cast<const bf16x8*>(src);
      if (kt + 32 < K)  // warm next K-tile (speculative, dropped if invalid)
        __builtin_prefetch(src + 32, 0, 1);
    }
    // ---- stage B tile (32 x 64) transposed into sBt[n][k]
    {
      bf16 tmp[8];
      const size_t rowBase = bOff + (size_t)(kt + bk) * ldb + cN + bn8;
      if (Bf) {
        if (fullN) {
          f32x4 v0, v1;  // ldb may be odd (50257): unaligned-safe vector loads
          __builtin_memcpy(&v0, Bf + rowBase, 16);
          __builtin_memcpy(&v1, Bf + rowBase + 4, 16);
          #pragma unroll
          for (int j = 0; j < 4; ++j) { tmp[j] = (bf16)v0[j]; tmp[4 + j] = (bf16)v1[j]; }
        } else {
          #pragma unroll
          for (int j = 0; j < 8; ++j) {
            const int col = cN + bn8 + j;
            tmp[j] = (col < N) ? (bf16)Bf[rowBase + j] : (bf16)0.f;
          }
        }
        if (kt + 32 < K)
          __builtin_prefetch(Bf + rowBase + (size_t)32 * ldb, 0, 1);
      } else {
        bf16x8 v;
        __builtin_memcpy(&v, Bb + rowBase, 16);
        #pragma unroll
        for (int j = 0; j < 8; ++j) tmp[j] = v[j];
        if (kt + 32 < K)
          __builtin_prefetch(Bb + rowBase + (size_t)32 * ldb, 0, 1);
      }
      #pragma unroll
      for (int j = 0; j < 8; ++j) sBt[bn8 + j][bk] = tmp[j];
    }
    __syncthreads();

    // B fragments: lane -> N = nw*32 + j*16 + l16; elems e: K = half*16 + e
    FragBF bf0, bf1;
    bf0.h[0] = *reinterpret_cast<const bf16x8*>(&sBt[nw * 32 + l16][half * 16]);
    bf0.h[1] = *reinterpret_cast<const bf16x8*>(&sBt[nw * 32 + l16][half * 16 + 8]);
    bf1.h[0] = *reinterpret_cast<const bf16x8*>(&sBt[nw * 32 + 16 + l16][half * 16]);
    bf1.h[1] = *reinterpret_cast<const bf16x8*>(&sBt[nw * 32 + 16 + l16][half * 16 + 8]);

    // A fragments: lane -> M = base + l16;
    //   elems e: K = half*16 + e
    FragBF a0, a1;
    a0.h[0] = *reinterpret_cast<const bf16x8*>(&sA[mw * 32 + l16][half * 16]);
    a0.h[1] = *reinterpret_cast<const bf16x8*>(&sA[mw * 32 + l16][half * 16 + 8]);
    a1.h[0] = *reinterpret_cast<const bf16x8*>(&sA[mw * 32 + 16 + l16][half * 16]);
    a1.h[1] = *reinterpret_cast<const bf16x8*>(&sA[mw * 32 + 16 + l16][half * 16 + 8]);

    acc[0][0] = __builtin_amdgcn_wmma_f32_16x16x32_bf16(false, a0.v, false, bf0.v,
                                                        (short)0, acc[0][0], false, false);
    acc[0][1] = __builtin_amdgcn_wmma_f32_16x16x32_bf16(false, a0.v, false, bf1.v,
                                                        (short)0, acc[0][1], false, false);
    acc[1][0] = __builtin_amdgcn_wmma_f32_16x16x32_bf16(false, a1.v, false, bf0.v,
                                                        (short)0, acc[1][0], false, false);
    acc[1][1] = __builtin_amdgcn_wmma_f32_16x16x32_bf16(false, a1.v, false, bf1.v,
                                                        (short)0, acc[1][1], false, false);
    __syncthreads();
  }

  // ---- epilogue: C/D layout (ISA): VGPR r -> M = r + 8*half, N = l16
  #pragma unroll
  for (int j = 0; j < 2; ++j) {
    const int col = cN + nw * 32 + j * 16 + l16;
    if (col < N) {
      const float bv = bias ? bias[col] : 0.f;
      #pragma unroll
      for (int i = 0; i < 2; ++i) {
        #pragma unroll
        for (int r = 0; r < 8; ++r) {
          const int row = cM + mw * 32 + i * 16 + half * 8 + r;
          float v = acc[i][j][r] + bv;
          if (relu) v = v > 0.f ? v : 0.f;
          const size_t idx = cOff + (size_t)row * ldc + col;
          if (resid) v += resid[idx];
          if (Cf) Cf[idx] = v;
          if (Cb) Cb[idx] = (bf16)v;
        }
      }
    }
  }
}

// ---------------------------------------------------------------------------
// x[b,t,:] = tok_emb[idx[b,t],:] + pos_emb[t,:]
// ---------------------------------------------------------------------------
__global__ __launch_bounds__(256) void embed_kernel(
    const int* __restrict__ idx, const float* __restrict__ tok,
    const float* __restrict__ pos, float* __restrict__ x)
{
  const int i = blockIdx.x * 256 + threadIdx.x;
  if (i >= GPT_R * GPT_D) return;
  const int row = i >> 10;
  const int d   = i & 1023;
  const int token = idx[row];
  x[i] = tok[(size_t)token * GPT_D + d] +
         pos[(size_t)(row & (GPT_T - 1)) * GPT_D + d];
}

// ---------------------------------------------------------------------------
// LayerNorm over D=1024, output bf16 (feeds WMMA A operand). 1 block / row.
// ---------------------------------------------------------------------------
__global__ __launch_bounds__(256) void layernorm_to_bf16(
    const float* __restrict__ x, const float* __restrict__ g,
    const float* __restrict__ b, bf16* __restrict__ out)
{
  __shared__ float sh[256];
  const int row = blockIdx.x;
  const int tid = threadIdx.x;
  float v[4];
  float s = 0.f;
  #pragma unroll
  for (int i = 0; i < 4; ++i) {
    v[i] = x[(size_t)row * GPT_D + tid + i * 256];
    s += v[i];
  }
  const float mean = blockReduceSum256(s, sh) * (1.f / GPT_D);
  float sq = 0.f;
  #pragma unroll
  for (int i = 0; i < 4; ++i) {
    const float d = v[i] - mean;
    sq += d * d;
  }
  const float var  = blockReduceSum256(sq, sh) * (1.f / GPT_D);
  const float rstd = rsqrtf(var + 1e-5f);
  #pragma unroll
  for (int i = 0; i < 4; ++i) {
    const int d = tid + i * 256;
    out[(size_t)row * GPT_D + d] = (bf16)((v[i] - mean) * rstd * g[d] + b[d]);
  }
}

// ---------------------------------------------------------------------------
// Pack Wq/Wk/Wv[l] ([H,D,HS] each, f32) -> bf16 row-major [D, 3*D]
// out column c: which=c/1024, h=(c%1024)/64, s=c%64
// ---------------------------------------------------------------------------
__global__ __launch_bounds__(256) void permute_qkv_w(
    const float* __restrict__ Wq, const float* __restrict__ Wk,
    const float* __restrict__ Wv, bf16* __restrict__ out)
{
  const int i = blockIdx.x * 256 + threadIdx.x;
  if (i >= GPT_D * 3 * GPT_D) return;
  const int d  = i / (3 * GPT_D);
  const int c  = i % (3 * GPT_D);
  const int wh = c >> 10;
  const int cc = c & 1023;
  const int h  = cc >> 6;
  const int s  = cc & 63;
  const float* W = (wh == 0) ? Wq : (wh == 1) ? Wk : Wv;
  out[i] = (bf16)W[((size_t)h * GPT_D + d) * GPT_HS + s];
}

// ---------------------------------------------------------------------------
// qkv f32 [R, 3*D] -> per-head bf16:
//   qb [bh][t][s], kTb [bh][s][t] (K transposed for scores GEMM), vb [bh][t][s]
// ---------------------------------------------------------------------------
__global__ __launch_bounds__(256) void pack_qkv(
    const float* __restrict__ qkv, bf16* __restrict__ qb,
    bf16* __restrict__ kTb, bf16* __restrict__ vb)
{
  const int i = blockIdx.x * 256 + threadIdx.x;
  if (i >= GPT_R * GPT_D) return;
  const int row = i >> 10;           // b*T + t
  const int c   = i & 1023;          // h*64 + s
  const int h   = c >> 6;
  const int s   = c & 63;
  const int b   = row >> 8;
  const int t   = row & 255;
  const size_t bh   = (size_t)b * GPT_H + h;
  const size_t base = (size_t)row * (3 * GPT_D);
  qb [(bh * GPT_T + t) * GPT_HS + s] = (bf16)qkv[base + c];
  kTb[(bh * GPT_HS + s) * GPT_T + t] = (bf16)qkv[base + GPT_D + c];
  vb [(bh * GPT_T + t) * GPT_HS + s] = (bf16)qkv[base + 2 * GPT_D + c];
}

// ---------------------------------------------------------------------------
// Causal softmax over one row of scores (scale 1/sqrt(64)); bf16 probs out.
// grid = (T, B*H); block = T threads.
// ---------------------------------------------------------------------------
__global__ __launch_bounds__(256) void softmax_causal(
    const float* __restrict__ scores, bf16* __restrict__ probs)
{
  __shared__ float sh[256];
  const int t = blockIdx.x;
  const int z = blockIdx.y;
  const int u = threadIdx.x;
  const size_t base = ((size_t)z * GPT_T + t) * GPT_T;
  const bool valid = (u <= t);
  const float raw = scores[base + u];
  const float v = valid ? raw * 0.125f : -3.0e38f;
  const float mx = blockReduceMax256(v, sh);
  const float e = valid ? __expf(v - mx) : 0.f;
  const float sum = blockReduceSum256(e, sh);
  probs[base + u] = (bf16)(e / sum);
}

// ---------------------------------------------------------------------------
// Per-row NLL from logits, then deterministic mean reduction.
// ---------------------------------------------------------------------------
__global__ __launch_bounds__(256) void loss_rows(
    const float* __restrict__ logits, const int* __restrict__ targets,
    float* __restrict__ rowloss)
{
  __shared__ float sh[256];
  const int row = blockIdx.x;
  const int tid = threadIdx.x;
  const float* lr = logits + (size_t)row * GPT_V;
  float m = -3.0e38f;
  for (int j = tid; j < GPT_V; j += 256) m = fmaxf(m, lr[j]);
  m = blockReduceMax256(m, sh);
  float s = 0.f;
  for (int j = tid; j < GPT_V; j += 256) s += __expf(lr[j] - m);
  s = blockReduceSum256(s, sh);
  if (tid == 0) {
    const int tgt = targets[row];
    rowloss[row] = -(lr[tgt] - m - __logf(s));
  }
}

__global__ __launch_bounds__(256) void loss_reduce(
    const float* __restrict__ rowloss, float* __restrict__ out)
{
  __shared__ float sh[256];
  float s = 0.f;
  for (int j = threadIdx.x; j < GPT_R; j += 256) s += rowloss[j];
  s = blockReduceSum256(s, sh);
  if (threadIdx.x == 0) *out = s * (1.f / GPT_R);
}

// ---------------------------------------------------------------------------
// Host side
// ---------------------------------------------------------------------------
static void launch_gemm(hipStream_t stream,
                        const bf16* A, long lda, long sAz,
                        const bf16* Bb, const float* Bf, long ldb, long sBz,
                        float* Cf, bf16* Cb, long ldc, long sC0, long sC1,
                        int zdiv, const float* bias, const float* resid,
                        int M, int N, int K, int relu, int batch)
{
  dim3 grid((N + 63) / 64, M / 128, batch);
  gemm_bf16_wmma<<<grid, 256, 0, stream>>>(A, lda, sAz, Bb, Bf, ldb, sBz,
                                           Cf, Cb, ldc, sC0, sC1, zdiv,
                                           bias, resid, M, N, K, relu);
}

extern "C" void kernel_launch(void* const* d_in, const int* in_sizes, int n_in,
                              void* d_out, int out_size, void* d_ws, size_t ws_size,
                              hipStream_t stream)
{
  (void)in_sizes; (void)n_in; (void)out_size; (void)ws_size;

  const int*   idx  = (const int*)  d_in[0];
  const int*   tgts = (const int*)  d_in[1];
  const float* tok  = (const float*)d_in[2];
  const float* pos  = (const float*)d_in[3];
  const float* Wq   = (const float*)d_in[4];
  const float* Wk   = (const float*)d_in[5];
  const float* Wv   = (const float*)d_in[6];
  const float* ln1g = (const float*)d_in[7];
  const float* ln1b = (const float*)d_in[8];
  const float* ln2g = (const float*)d_in[9];
  const float* ln2b = (const float*)d_in[10];
  const float* fc1w = (const float*)d_in[11];
  const float* fc1b = (const float*)d_in[12];
  const float* fc2w = (const float*)d_in[13];
  const float* fc2b = (const float*)d_in[14];
  const float* lnfg = (const float*)d_in[15];
  const float* lnfb = (const float*)d_in[16];
  const float* lmw  = (const float*)d_in[17];
  const float* lmb  = (const float*)d_in[18];

  float* logits  = (float*)d_out;
  float* lossOut = logits + (size_t)GPT_R * GPT_V;

  // workspace carve-up (~124 MB total)
  char*  ws  = (char*)d_ws;
  size_t off = 0;
  auto alloc = [&](size_t bytes) -> void* {
    void* p = ws + off;
    off += (bytes + 255) & ~(size_t)255;
    return p;
  };
  float* x      = (float*)alloc((size_t)GPT_R * GPT_D * 4);
  bf16*  hb     = (bf16*) alloc((size_t)GPT_R * GPT_D * 2);
  bf16*  wqkv   = (bf16*) alloc((size_t)GPT_D * 3 * GPT_D * 2);
  float* qkv    = (float*)alloc((size_t)GPT_R * 3 * GPT_D * 4);
  bf16*  qb     = (bf16*) alloc((size_t)GPT_R * GPT_D * 2);
  bf16*  kTb    = (bf16*) alloc((size_t)GPT_R * GPT_D * 2);
  bf16*  vb     = (bf16*) alloc((size_t)GPT_R * GPT_D * 2);
  float* scores = (float*)alloc((size_t)GPT_B * GPT_H * GPT_T * GPT_T * 4);
  bf16*  probs  = (bf16*) alloc((size_t)GPT_B * GPT_H * GPT_T * GPT_T * 2);
  bf16*  ff1b   = (bf16*) alloc((size_t)GPT_R * 4 * GPT_D * 2);
  float* rowls  = (float*)alloc((size_t)GPT_R * 4);

  const int BH = GPT_B * GPT_H;                 // 128 attention batches
  const long sScore = (long)GPT_T * GPT_T;      // 65536
  const long sHead  = (long)GPT_T * GPT_HS;     // 16384

  // 1) embeddings
  embed_kernel<<<(GPT_R * GPT_D) / 256, 256, 0, stream>>>(idx, tok, pos, x);

  for (int l = 0; l < GPT_L; ++l) {
    const float* Wql = Wq + (size_t)l * GPT_H * GPT_D * GPT_HS;
    const float* Wkl = Wk + (size_t)l * GPT_H * GPT_D * GPT_HS;
    const float* Wvl = Wv + (size_t)l * GPT_H * GPT_D * GPT_HS;

    // 2) h = LN1(x) -> bf16
    layernorm_to_bf16<<<GPT_R, 256, 0, stream>>>(
        x, ln1g + (size_t)l * GPT_D, ln1b + (size_t)l * GPT_D, hb);

    // 3) pack QKV weights [D, 3D] bf16
    permute_qkv_w<<<(GPT_D * 3 * GPT_D) / 256, 256, 0, stream>>>(Wql, Wkl, Wvl, wqkv);

    // 4) qkv = h @ Wqkv   (2048 x 3072, K=1024)
    launch_gemm(stream, hb, GPT_D, 0, wqkv, nullptr, 3 * GPT_D, 0,
                qkv, nullptr, 3 * GPT_D, 0, 0, 1, nullptr, nullptr,
                GPT_R, 3 * GPT_D, GPT_D, 0, 1);

    // 5) split into per-head bf16 q, k^T, v
    pack_qkv<<<(GPT_R * GPT_D) / 256, 256, 0, stream>>>(qkv, qb, kTb, vb);

    // 6) scores[z] = q[z] @ k[z]^T   (128 x [256x256, K=64])
    launch_gemm(stream, qb, GPT_HS, sHead, kTb, nullptr, GPT_T, sHead,
                scores, nullptr, GPT_T, sScore, 0, 1, nullptr, nullptr,
                GPT_T, GPT_T, GPT_HS, 0, BH);

    // 7) causal softmax (scale 1/8) -> bf16 probs
    softmax_causal<<<dim3(GPT_T, BH), 256, 0, stream>>>(scores, probs);

    // 8) x += probs @ v, heads written interleaved into x:
    //    C offset for z=(b*H+h): b*(T*D) + h*64
    launch_gemm(stream, probs, GPT_T, sScore, vb, nullptr, GPT_HS, sHead,
                x, nullptr, GPT_D, (long)GPT_T * GPT_D, GPT_HS, GPT_H,
                nullptr, x, GPT_T, GPT_HS, GPT_T, 0, BH);

    // 9) h = LN2(x) -> bf16
    layernorm_to_bf16<<<GPT_R, 256, 0, stream>>>(
        x, ln2g + (size_t)l * GPT_D, ln2b + (size_t)l * GPT_D, hb);

    // 10) ff1 = relu(h @ fc1_w + fc1_b) -> bf16  (2048 x 4096, K=1024)
    launch_gemm(stream, hb, GPT_D, 0,
                nullptr, fc1w + (size_t)l * GPT_D * 4 * GPT_D, 4 * GPT_D, 0,
                nullptr, ff1b, 4 * GPT_D, 0, 0, 1,
                fc1b + (size_t)l * 4 * GPT_D, nullptr,
                GPT_R, 4 * GPT_D, GPT_D, 1, 1);

    // 11) x += ff1 @ fc2_w + fc2_b   (2048 x 1024, K=4096)
    launch_gemm(stream, ff1b, 4 * GPT_D, 0,
                nullptr, fc2w + (size_t)l * 4 * GPT_D * GPT_D, GPT_D, 0,
                x, nullptr, GPT_D, 0, 0, 1,
                fc2b + (size_t)l * GPT_D, x,
                GPT_R, GPT_D, 4 * GPT_D, 0, 1);
  }

  // 12) final LN
  layernorm_to_bf16<<<GPT_R, 256, 0, stream>>>(x, lnfg, lnfb, hb);

  // 13) logits = h @ lm_w + lm_b   (2048 x 50257, K=1024; N bounds-checked)
  launch_gemm(stream, hb, GPT_D, 0, nullptr, lmw, GPT_V, 0,
              logits, nullptr, GPT_V, 0, 0, 1, lmb, nullptr,
              GPT_R, GPT_V, GPT_D, 0, 1);

  // 14) loss = mean NLL (deterministic two-stage reduction)
  loss_rows<<<GPT_R, 256, 0, stream>>>(logits, tgts, rowls);
  loss_reduce<<<1, 256, 0, stream>>>(rowls, lossOut);
}
